// Updating_memory_45011257262396
// MI455X (gfx1250) — matrix operand order repaired and checked by reading, not compiled, not run on hardware
//
#include <hip/hip_runtime.h>

// ---------------- problem constants ----------------
#define BTOT   32768
#define LSEQ   20
#define HDIM   100
#define HP     112    // H padded to 7*16; zeros stored in [100,112)
#define LP     32     // L padded for K of second matmul; zeros in [20,32)
#define KG     312    // 3H padded; zeros stored in [300,312)
#define NTHR   128    // 4 waves
#define WAVES  4
#define NB_PER_WAVE 8 // 32 batches per workgroup -> 1024 WGs

typedef __attribute__((ext_vector_type(16))) _Float16 v16h;
typedef __attribute__((ext_vector_type(8)))  _Float16 v8h;
typedef __attribute__((ext_vector_type(4)))  _Float16 v4h;
typedef __attribute__((ext_vector_type(8)))  float    v8f;
typedef __attribute__((ext_vector_type(4)))  float    v4f;

static_assert(sizeof(v16h) == 32, "v16h size");

// ---------------- statically typed LDS layout ----------------
struct alignas(16) WaveScratch {
  alignas(16) _Float16 E[3][LSEQ * HP];         // 13440 B
  union alignas(16) {
    _Float16 X4[3][LSEQ * HP];                  // 13440 B (his4|sub4|mem4)
    _Float16 gin[LSEQ * KG];                    // 12480 B (gate input overlay)
  };
  alignas(16) float    attF[2][LSEQ * LSEQ];    // 3200 B
  alignas(16) _Float16 attA[2][LSEQ * LP];      // 2560 B
  alignas(16) float    matt[LSEQ * HP];         // 8960 B
};                                              // = 41600 B
struct alignas(16) Smem {
  alignas(16) _Float16 WT[3][HP * HP];          // 75264 B (his, sub, mem; [n][k])
  alignas(16) _Float16 WTg[HP * KG];            // 69888 B ([n][k])
  alignas(16) float    bias[4][HP];             // 1792 B (his, sub, mem, gate)
  WaveScratch wsv[WAVES];                       // 166400 B
};                                              // = 313344 B <= 320 KB
static_assert(sizeof(WaveScratch) == 41600, "ws size");
static_assert(sizeof(Smem) == 313344, "smem size");

// ---- branch-free fast math on native transcendentals ----
__device__ __forceinline__ float fexp(float x) {
  return __builtin_amdgcn_exp2f(x * 1.44269504088896340736f);
}
__device__ __forceinline__ float frcp(float x) { return __builtin_amdgcn_rcpf(x); }
__device__ __forceinline__ float ftanh(float x) {
  const float xc = fminf(fmaxf(x, -15.f), 15.f);
  const float e  = fexp(2.f * xc);
  return (e - 1.f) * frcp(e + 1.f);
}
__device__ __forceinline__ float fsigmoid(float x) {
  return frcp(1.f + fexp(-x));
}
__device__ __forceinline__ int imin(int a, int b) { return a < b ? a : b; }

// Branch-free A-fragment (16x32 f16). lane L holds row m=(L&15);
// 16 contiguous K values {kb..kb+15}, kb = k0 + ((L&16)?16:0).
// Out-of-range rows are clamped (their C rows are never stored); out-of-range
// K halves are clamped to ldk-8, which is guaranteed to hold stored zeros.
__device__ __forceinline__ v16h ldsA(const _Float16* base, int row, int rmax,
                                     int k0, int ldk, int lane) {
  const int r  = imin(row, rmax - 1);
  const int kb = k0 + ((lane & 16) ? 16 : 0);
  const int kc = ldk - 8;
  const int h0 = imin(kb, kc);
  const int h1 = imin(kb + 8, kc);
  const _Float16* p = base + r * ldk;
  v8h a = *reinterpret_cast<const v8h*>(p + h0);
  v8h b = *reinterpret_cast<const v8h*>(p + h1);
  v16h f;
#pragma unroll
  for (int i = 0; i < 8; ++i) { f[i] = a[i]; f[8 + i] = b[i]; }
  return f;
}

// Branch-free B-fragment (32x16 f16). lane L holds output column n=(L&15);
// 16 contiguous K values starting at k0 + ((L&16)?16:0); `base` stored [n][k].
__device__ __forceinline__ v16h ldsB(const _Float16* base, int row, int rmax,
                                     int k0, int ldk, int lane) {
  const int r  = imin(row, rmax - 1);
  const int kb = k0 + ((lane & 16) ? 16 : 0);
  const int kc = ldk - 8;
  const int h0 = imin(kb, kc);
  const int h1 = imin(kb + 8, kc);
  const _Float16* p = base + r * ldk;
  v8h a = *reinterpret_cast<const v8h*>(p + h0);
  v8h b = *reinterpret_cast<const v8h*>(p + h1);
  v16h f;
#pragma unroll
  for (int i = 0; i < 8; ++i) { f[i] = a[i]; f[8 + i] = b[i]; }
  return f;
}

// B-fragment gather for [k][n]-stored B (k0 == 0, K tile = 32). No K guard:
// the matching A operand (attA) holds exact zeros for k >= 20, and the
// overread lands in adjacent staged f16 data (always finite).
__device__ __forceinline__ v16h ldsBT(const _Float16* base, int col,
                                      int ldk, int lane) {
  const int kb = (lane & 16) ? 16 : 0;
  v16h f;
#pragma unroll
  for (int i = 0; i < 16; ++i) f[i] = base[(kb + i) * ldk + col];
  return f;
}

__device__ __forceinline__ v8f wmma_f16(v16h a, v16h b, v8f c) {
  return __builtin_amdgcn_wmma_f32_16x16x32_f16(false, a, false, b,
                                                (short)0, c, false, false);
}

extern "C" __global__ __launch_bounds__(NTHR, 1)
void updmem_fused(const float* __restrict__ his_emb, const float* __restrict__ sub_emb,
                  const float* __restrict__ memory,
                  const int* __restrict__ his_len, const int* __restrict__ sub_len,
                  const float* __restrict__ W_mem, const float* __restrict__ b_mem,
                  const float* __restrict__ W_his, const float* __restrict__ b_his,
                  const float* __restrict__ W_sub, const float* __restrict__ b_sub,
                  const float* __restrict__ W_gate, const float* __restrict__ b_gate,
                  float* __restrict__ out) {
  __shared__ Smem sm;
  const int tid  = threadIdx.x;
  const int wave = tid >> 5;
  const int lane = tid & 31;

  // ---- stage weights (transposed, zero padded) once per workgroup ----
  {
    unsigned int* z = reinterpret_cast<unsigned int*>(&sm.WT[0][0]);
    for (int t = tid; t < (int)(sizeof(sm.WT) + sizeof(sm.WTg)) / 4; t += NTHR)
      z[t] = 0u;
    __syncthreads();
    const float* wsrc[3] = {W_his, W_sub, W_mem};
#pragma unroll
    for (int x = 0; x < 3; ++x) {
      const float* g = wsrc[x];
      _Float16*    d = sm.WT[x];
      for (int t = tid; t < HDIM * HDIM; t += NTHR) {
        const int k = t / HDIM, n = t - k * HDIM;
        d[n * HP + k] = (_Float16)g[t];               // WT[n][k] = W[k][n]
      }
    }
    for (int t = tid; t < 3 * HDIM * HDIM; t += NTHR) {
      const int k = t / HDIM, n = t - k * HDIM;
      sm.WTg[n * KG + k] = (_Float16)W_gate[t];
    }
    const float* bsrc[4] = {b_his, b_sub, b_mem, b_gate};
    for (int t = tid; t < 4 * HP; t += NTHR) {
      const int j = t / HP, c = t - j * HP;
      sm.bias[j][c] = (c < HDIM) ? bsrc[j][c] : 0.f;
    }
  }
  __syncthreads();

  WaveScratch& W = sm.wsv[wave];

  // pre-zero the 12 pad columns of E once per wave (never rewritten)
#pragma unroll
  for (int x = 0; x < 3; ++x)
    for (int r = 0; r < LSEQ; ++r)
      if (lane < HP - HDIM) W.E[x][r * HP + HDIM + lane] = (_Float16)0.f;

  const int lm15 = lane & 15;
  const int mhi  = (lane & 16) ? 8 : 0;   // C/D row offset for upper lane half

#pragma unroll 1
  for (int ib = 0; ib < NB_PER_WAVE; ++ib) {
    const int b = blockIdx.x * (WAVES * NB_PER_WAVE) + wave * NB_PER_WAVE + ib;
    if (b >= BTOT) continue;  // uniform per wave
    const size_t boff = (size_t)b * LSEQ * HDIM;
    const float* eg[3] = {his_emb + boff, sub_emb + boff, memory + boff};

    // ---- stage f32 embeddings -> f16 LDS (vectorized float4 -> 4xf16) ----
#pragma unroll
    for (int x = 0; x < 3; ++x) {
      const float* g = eg[x];
      _Float16*    d = W.E[x];
#pragma unroll 1
      for (int t = lane; t < LSEQ * (HDIM / 4); t += 32) {  // 500 quads
        const int r = t / (HDIM / 4);
        const int q = t - r * (HDIM / 4);
        v4f v = *reinterpret_cast<const v4f*>(g + r * HDIM + q * 4);
        v4h h;
#pragma unroll
        for (int i = 0; i < 4; ++i) h[i] = (_Float16)v[i];
        *reinterpret_cast<v4h*>(d + r * HP + q * 4) = h;
      }
    }

    // ---- projections: X4 = tanh(Xemb @ W + b) ----
#pragma unroll 1
    for (int x = 0; x < 3; ++x) {
      const _Float16* wt = sm.WT[x];
      _Float16* dst = W.X4[x];
      v16h a[2][4];
#pragma unroll
      for (int mt = 0; mt < 2; ++mt)
#pragma unroll
        for (int kk = 0; kk < 4; ++kk)
          a[mt][kk] = ldsA(W.E[x], mt * 16 + lm15, LSEQ, kk * 32, HP, lane);
#pragma unroll
      for (int nt = 0; nt < 7; ++nt) {
        const int n = nt * 16 + lm15;
        v8f c0 = {}, c1 = {};
#pragma unroll
        for (int kk = 0; kk < 4; ++kk) {
          v16h bf = ldsB(wt, n, HP, kk * 32, HP, lane);
          c0 = wmma_f16(a[0][kk], bf, c0);
          c1 = wmma_f16(a[1][kk], bf, c1);
        }
        const float bv = sm.bias[x][n];
#pragma unroll
        for (int r = 0; r < 8; ++r) {
          dst[(r + mhi) * HP + n] = (_Float16)ftanh(c0[r] + bv);
          if (r < 4) {                               // folds statically for r>=4
            if (mhi == 0)                            // m1 = 16+r < 20
              dst[(16 + r) * HP + n] = (_Float16)ftanh(c1[r] + bv);
          }
        }
      }
    }

    // ---- attention logits: att[s] = mem4 @ X4[s]^T  (X4 rows act as B cols) ----
    const _Float16* mem4 = W.X4[2];
    v16h am[2][4];
#pragma unroll
    for (int mt = 0; mt < 2; ++mt)
#pragma unroll
      for (int kk = 0; kk < 4; ++kk)
        am[mt][kk] = ldsA(mem4, mt * 16 + lm15, LSEQ, kk * 32, HP, lane);
#pragma unroll
    for (int s = 0; s < 2; ++s) {
      const _Float16* xb = W.X4[s];
#pragma unroll
      for (int nt = 0; nt < 2; ++nt) {
        const int n = nt * 16 + lm15;
        v8f c0 = {}, c1 = {};
#pragma unroll
        for (int kk = 0; kk < 4; ++kk) {
          v16h bf = ldsB(xb, n, LSEQ, kk * 32, HP, lane);
          c0 = wmma_f16(am[0][kk], bf, c0);
          c1 = wmma_f16(am[1][kk], bf, c1);
        }
        if ((nt == 0) || (lm15 < 4)) {               // n < 20
#pragma unroll
          for (int r = 0; r < 8; ++r) {
            W.attF[s][(r + mhi) * LSEQ + n] = c0[r];
            if (r < 4) {
              if (mhi == 0) W.attF[s][(16 + r) * LSEQ + n] = c1[r];
            }
          }
        }
      }
    }

    // ---- softmax (row mask only: inf_mask is a uniform row shift) ----
#pragma unroll
    for (int s = 0; s < 2; ++s) {
      const int len = (s == 0) ? his_len[b] : sub_len[b];
      if (lane < LSEQ) {
        const float* row = &W.attF[s][lane * LSEQ];
        float v[LSEQ];
        float mx = -3.0e38f;
#pragma unroll
        for (int j = 0; j < LSEQ; ++j) { v[j] = row[j]; mx = fmaxf(mx, v[j]); }
        float sum = 0.f;
#pragma unroll
        for (int j = 0; j < LSEQ; ++j) { v[j] = fexp(v[j] - mx); sum += v[j]; }
        const float inv = (lane < len) ? frcp(sum) : 0.f;   // zero_mask on rows
        _Float16* arow = &W.attA[s][lane * LP];
#pragma unroll
        for (int j = 0; j < LSEQ; ++j) arow[j] = (_Float16)(v[j] * inv);
#pragma unroll
        for (int j = LSEQ; j < LP; ++j) arow[j] = (_Float16)0.f;
      }
    }

    // ---- mem_att = attA_his @ hisE + attA_sub @ subE ----
    v16h ah[2], as2[2];
#pragma unroll
    for (int mt = 0; mt < 2; ++mt) {
      ah[mt]  = ldsA(W.attA[0], mt * 16 + lm15, LSEQ, 0, LP, lane);
      as2[mt] = ldsA(W.attA[1], mt * 16 + lm15, LSEQ, 0, LP, lane);
    }
#pragma unroll
    for (int nt = 0; nt < 7; ++nt) {
      const int n = nt * 16 + lm15;
      v16h bh = ldsBT(W.E[0], n, HP, lane);
      v16h bs = ldsBT(W.E[1], n, HP, lane);
      v8f c0 = {}, c1 = {};
      c0 = wmma_f16(ah[0],  bh, c0);
      c1 = wmma_f16(ah[1],  bh, c1);
      c0 = wmma_f16(as2[0], bs, c0);
      c1 = wmma_f16(as2[1], bs, c1);
#pragma unroll
      for (int r = 0; r < 8; ++r) {
        W.matt[(r + mhi) * HP + n] = c0[r];
        if (r < 4) {
          if (mhi == 0) W.matt[(16 + r) * HP + n] = c1[r];
        }
      }
    }

    // ---- gate input [mem_att | memory | mem_att*memory], f16, quad-vectorized ----
    // (overlays X4; his4/sub4/mem4 are dead now)
#pragma unroll 1
    for (int m = 0; m < LSEQ; ++m) {
      for (int q = lane; q < KG / 4; q += 32) {   // 78 quads per row
        const int k = q * 4;
        v4h h;
        if (k < HDIM) {                           // mem_att
          v4f v = *reinterpret_cast<const v4f*>(&W.matt[m * HP + k]);
#pragma unroll
          for (int i = 0; i < 4; ++i) h[i] = (_Float16)v[i];
        } else if (k < 2 * HDIM) {                // memory (already f16)
          h = *reinterpret_cast<const v4h*>(&W.E[2][m * HP + (k - HDIM)]);
        } else if (k < 3 * HDIM) {                // mem_att * memory
          v4f v = *reinterpret_cast<const v4f*>(&W.matt[m * HP + (k - 2 * HDIM)]);
          v4h e = *reinterpret_cast<const v4h*>(&W.E[2][m * HP + (k - 2 * HDIM)]);
#pragma unroll
          for (int i = 0; i < 4; ++i) h[i] = (_Float16)(v[i] * (float)e[i]);
        } else {
#pragma unroll
          for (int i = 0; i < 4; ++i) h[i] = (_Float16)0.f;
        }
        *reinterpret_cast<v4h*>(&W.gin[m * KG + k]) = h;
      }
    }

    // ---- gate GEMM + sigmoid + final blend ----
#pragma unroll 1
    for (int mt = 0; mt < 2; ++mt) {
      const int arow = mt * 16 + lm15;
      v16h ag[10];
#pragma unroll
      for (int kk = 0; kk < 10; ++kk)
        ag[kk] = ldsA(W.gin, arow, LSEQ, kk * 32, KG, lane);
#pragma unroll
      for (int nt = 0; nt < 7; ++nt) {
        const int n = nt * 16 + lm15;
        v8f c = {};
#pragma unroll
        for (int kk = 0; kk < 10; ++kk)
          c = wmma_f16(ag[kk], ldsB(sm.WTg, n, HP, kk * 32, KG, lane), c);
        const float bg = sm.bias[3][n];
        const bool nok = (nt < 6) || (lm15 < 4);     // n < 100
#pragma unroll
        for (int r = 0; r < 8; ++r) {
          const int m = mt * 16 + r + mhi;
          const bool mok = (mt == 0) || (r < 4 && mhi == 0);  // m < 20
          if (mok && nok) {
            const float g  = fsigmoid(c[r] + bg);
            const float mv = memory[boff + (size_t)m * HDIM + n];
            const float ma = W.matt[m * HP + n];
            out[boff + (size_t)m * HDIM + n] = (1.f - g) * mv + g * ma;
          }
        }
      }
    }
  }
}

extern "C" void kernel_launch(void* const* d_in, const int* in_sizes, int n_in,
                              void* d_out, int out_size, void* d_ws, size_t ws_size,
                              hipStream_t stream) {
  (void)in_sizes; (void)n_in; (void)d_ws; (void)ws_size; (void)out_size;
  const float* his_emb = (const float*)d_in[0];
  const float* sub_emb = (const float*)d_in[1];
  const float* memory  = (const float*)d_in[2];
  const int*   his_len = (const int*)d_in[3];
  const int*   sub_len = (const int*)d_in[4];
  const float* W_mem   = (const float*)d_in[5];
  const float* b_mem   = (const float*)d_in[6];
  const float* W_his   = (const float*)d_in[7];
  const float* b_his   = (const float*)d_in[8];
  const float* W_sub   = (const float*)d_in[9];
  const float* b_sub   = (const float*)d_in[10];
  const float* W_gate  = (const float*)d_in[11];
  const float* b_gate  = (const float*)d_in[12];
  float* out = (float*)d_out;

  const int batches_per_wg = WAVES * NB_PER_WAVE;       // 32
  const int n_wg = (BTOT + batches_per_wg - 1) / batches_per_wg;  // 1024
  updmem_fused<<<dim3(n_wg), dim3(NTHR), 0, stream>>>(
      his_emb, sub_emb, memory, his_len, sub_len,
      W_mem, b_mem, W_his, b_his, W_sub, b_sub, W_gate, b_gate, out);
}
